// BinaryConv2d_89429809037361
// MI455X (gfx1250) — 8-bit matrix operand order repaired and checked by reading, compiled, not run on hardware
//
#include <hip/hip_runtime.h>


typedef __attribute__((ext_vector_type(8))) int v8i;
typedef __attribute__((ext_vector_type(4))) int v4i;

typedef __attribute__((address_space(1))) v4i* gv4p;
typedef __attribute__((address_space(3))) v4i* lv4p;

#define BATCH 32
#define CIN   64
#define COUT  64
#define HH    160
#define WW    160
#define NT    32
#define HALOW (NT + 2)

#if __has_builtin(__builtin_amdgcn_global_load_async_to_lds_b128)
#define HAVE_ASYNC_LDS 1
#else
#define HAVE_ASYNC_LDS 0
#endif

__global__ __launch_bounds__(64) void prep_w(const float* __restrict__ w,
                                             signed char* __restrict__ wq,
                                             float* __restrict__ scales) {
    int cout = blockIdx.x;
    int t = threadIdx.x;
    __shared__ float red[64];

    const float* wr = w + (size_t)cout * 576;
    float s = 0.f;
    for (int j = 0; j < 9; ++j) {
        float v = wr[t * 9 + j];
        v = fminf(fmaxf(v, -1.f), 1.f);
        s += fabsf(v);
    }
    red[t] = s;
    __syncthreads();
    for (int off = 32; off > 0; off >>= 1) {
        if (t < off) red[t] += red[t + off];
        __syncthreads();
    }
    if (t == 0) scales[cout] = red[0] * (1.0f / 576.0f);

    int mt = cout >> 4;
    int mm = cout & 15;
    int kk = t;
    int v  = ((kk >> 5) << 2) | ((kk >> 2) & 3);
    int hi = (kk >> 4) & 1;
    int bb = kk & 3;
    int lane = mm + hi * 16;
    size_t byteoff = (size_t)(lane * 8 + v) * 4 + bb;
    for (int kh = 0; kh < 3; ++kh)
        for (int kw = 0; kw < 3; ++kw) {
            int kt = kh * 3 + kw;
            float wv = wr[t * 9 + kh * 3 + kw];
            signed char sgn = (wv > 0.f) ? 1 : ((wv < 0.f) ? -1 : 0);
            wq[(size_t)(mt * 9 + kt) * 1024 + byteoff] = sgn;
        }
}

__global__ __launch_bounds__(256) void binarize_t(const float* __restrict__ x,
                                                  signed char* __restrict__ xbT) {
    int tx = threadIdx.x & 15;
    int ty = threadIdx.x >> 4;
    size_t pix = (size_t)blockIdx.x * 16 + tx;
    size_t bb  = pix / (HH * WW);
    size_t hw  = pix - bb * (HH * WW);
    const float* src = x + (bb * CIN + (size_t)ty * 4) * (HH * WW) + hw;
    int p = 0;
    #pragma unroll
    for (int j = 0; j < 4; ++j) {
        float v = src[(size_t)j * HH * WW];
        int s = (v > 0.f) ? 1 : ((v < 0.f) ? -1 : 0);
        p |= (s & 0xff) << (8 * j);
    }
    ((int*)xbT)[pix * (CIN / 4) + ty] = p;
}

__global__ __launch_bounds__(256) void bconv(const signed char* __restrict__ xbT,
                                             const signed char* __restrict__ wq,
                                             const float* __restrict__ scales,
                                             float* __restrict__ out) {
    __shared__ __align__(16) signed char ldsX[3 * HALOW * CIN];

    int w0 = blockIdx.x * NT;
    int h  = blockIdx.y;
    int b  = blockIdx.z;
    int t  = threadIdx.x;

    const int CHUNKS_PER_ROW = HALOW * 4;
    for (int i = t; i < 3 * CHUNKS_PER_ROW; i += 256) {
        int r = i / CHUNKS_PER_ROW;
        int c = i - r * CHUNKS_PER_ROW;
        int p = c >> 2;
        int o = (c & 3) << 4;
        int hs = h + r - 1;
        int wx = w0 - 1 + p;
        signed char* dst = &ldsX[(r * HALOW + p) * CIN + o];
        if (hs >= 0 && hs < HH && wx >= 0 && wx < WW) {
            const signed char* srcp =
                xbT + (((size_t)b * HH + hs) * WW + wx) * CIN + o;
#if HAVE_ASYNC_LDS
            __builtin_amdgcn_global_load_async_to_lds_b128(
                (gv4p)srcp, (lv4p)dst,  0,  0);
#else
            *(v4i*)dst = *(const v4i*)srcp;
#endif
        } else {
            v4i z = {};
            *(v4i*)dst = z;
        }
    }
#if HAVE_ASYNC_LDS
#if __has_builtin(__builtin_amdgcn_s_wait_asynccnt)
    __builtin_amdgcn_s_wait_asynccnt(0);
#else
    asm volatile("s_wait_asynccnt 0x0" ::: "memory");
#endif
#endif
    __syncthreads();

    int lane = t & 31;
    int wave = t >> 5;
    int mt = wave >> 1;
    int nt = wave & 1;

    int colp = nt * 16 + (lane & 15);
    int kb0  = (lane >> 4) << 4;
    const signed char* bbase = &ldsX[colp * CIN + kb0];
    const signed char* aT = wq + (size_t)mt * 9 * 1024 + (size_t)lane * 32;

    v8i acc = {};
    #pragma unroll
    for (int kt = 0; kt < 9; ++kt) {
        int kh = kt / 3;
        int kw = kt - kh * 3;
        int boff = (kh * HALOW + kw) * CIN;
        v4i a0 = *(const v4i*)(aT + kt * 1024);
        v4i a1 = *(const v4i*)(aT + kt * 1024 + 16);
        v4i b0 = *(const v4i*)(bbase + boff);
        v4i b1 = *(const v4i*)(bbase + boff + 32);
        v8i A = {a0[0], a0[1], a0[2], a0[3], a1[0], a1[1], a1[2], a1[3]};
        v8i B = {b0[0], b0[1], b0[2], b0[3], b1[0], b1[1], b1[2], b1[3]};
        acc = __builtin_amdgcn_wmma_i32_16x16x64_iu8(true, A, true, B, acc,
                                                     false, false);
    }

    int N  = lane & 15;
    int hi = lane >> 4;
    int wp = w0 + nt * 16 + N;
    #pragma unroll
    for (int r = 0; r < 8; ++r) {
        int cout = mt * 16 + r + hi * 8;
        float sc = scales[cout];
        out[(((size_t)b * COUT + cout) * HH + h) * WW + wp] = (float)acc[r] * sc;
    }
}

extern "C" void kernel_launch(void* const* d_in, const int* in_sizes, int n_in,
                              void* d_out, int out_size, void* d_ws, size_t ws_size,
                              hipStream_t stream) {
    const float* x = (const float*)d_in[0];
    const float* w = (const float*)d_in[1];
    float* out = (float*)d_out;

    float*       scales = (float*)d_ws;
    signed char* wq     = (signed char*)d_ws + 256;
    signed char* xbT    = (signed char*)d_ws + 40960;

    prep_w<<<COUT, 64, 0, stream>>>(w, wq, scales);

    int npix_blocks = (BATCH * HH * WW) / 16;
    binarize_t<<<npix_blocks, 256, 0, stream>>>(x, xbT);

    dim3 grid(WW / NT, HH, BATCH);
    bconv<<<grid, 256, 0, stream>>>(xbT, wq, scales, out);
}
